// KimiK25TextModel_56642028700017
// MI455X (gfx1250) — compile-verified, matrix operand order checked by reading, not run on hardware
//
#include <hip/hip_runtime.h>
#include <hip/hip_bf16.h>

// ---------------- model constants ----------------
#define S_LEN   1024
#define DMODEL  2048
#define NHEADS  16
#define QLORA   768
#define KVLORA  512
#define ND_     128
#define RD_     64
#define VD_     128
#define QHD_    192      // ND_ + RD_
#define I_FF    8192
#define NEXP    16
#define MI_     512
#define SI_     512
#define VOCAB   32000
#define TOPK_   4
#define NGRP    4
#define RSF_    2.5f
#define EPS_    1e-6f

// Flip to 0 if the assembler rejects the async-to-LDS mnemonic.
#ifndef USE_ASYNC_LDS
#define USE_ASYNC_LDS 1
#endif

typedef __attribute__((ext_vector_type(16))) _Float16 v16h;
typedef __attribute__((ext_vector_type(8)))  _Float16 v8h;
typedef __attribute__((ext_vector_type(8)))  float    v8f;

// ---- CDNA5 async global->LDS copy (16B per lane), ISA §15.18.3 op 98 ----
__device__ __forceinline__ void async_copy_b128(unsigned ldsOff, const void* gsrc) {
#if USE_ASYNC_LDS
  asm volatile("global_load_async_to_lds_b128 %0, %1, off"
               :: "v"(ldsOff), "v"(gsrc) : "memory");
#else
  (void)ldsOff; (void)gsrc;
#endif
}
__device__ __forceinline__ void wait_async0() {
#if defined(__has_builtin)
#if __has_builtin(__builtin_amdgcn_s_wait_asynccnt)
  __builtin_amdgcn_s_wait_asynccnt(0);
#else
  asm volatile("s_wait_asynccnt 0x0" ::: "memory");
#endif
#else
  asm volatile("s_wait_asynccnt 0x0" ::: "memory");
#endif
}

// =====================================================================
// WMMA GEMM workhorse.
//   C[m,n] = (ACC ? C[m,n] : 0) + sum_k A[m,k] * (TRANSB ? B[n,k] : B[k,n])
// A is fp32 (converted to f16 while staging) or, when AHALF, native f16
// staged with GLOBAL_LOAD_ASYNC_TO_LDS_B128 (no conversion needed).
// Batched over blockIdx.z via sA/sB/sC element strides.
// Segmented mode (seg != null): M = seg[z+1]-seg[z]; A row m -> gat[seg[z]+m]
// (or seg[z]+m if gat==null); C row = seg[z]+m.  Used for MoE expert GEMMs.
// Block tile 128x128, K step 32, 8 waves, each wave 2x4 tiles of 16x16.
// AHALF path assumes full tiles (M%128==0, K%32==0) -- used for attention AV.
// =====================================================================
struct GemmArgs {
  const void*  A; int lda; long long sA;   // lda/sA in A-elements
  const float* B; int ldb; long long sB;
  float*       C; int ldc; long long sC;
  int M, N, K;
  const int* seg;   // optional [batch+1]
  const int* gat;   // optional row gather (token ids)
};

template<bool TRANSB, bool ACC, bool AHALF>
__global__ __launch_bounds__(256) void k_gemm(GemmArgs g)
{
  const int z = blockIdx.z;
  const float* B = g.B + (long long)z * g.sB;
  float*       C = g.C + (long long)z * g.sC;

  int M = g.M;
  int rowBase = 0;
  const int* gat = nullptr;
  if (g.seg) {
    rowBase = g.seg[z];
    M = g.seg[z + 1] - rowBase;
    gat = g.gat ? (g.gat + rowBase) : nullptr;
  }
  const int m0 = blockIdx.y * 128;
  const int n0 = blockIdx.x * 128;
  if (m0 >= M) return;   // uniform early-exit (empty expert tile)

  __shared__ __align__(16) _Float16 lsA[128 * 40];   // padded: 40 halves/row
  __shared__ __align__(16) _Float16 lsB[128 * 40];

  const int tid  = threadIdx.x;
  const int lane = tid & 31;
  const int wave = tid >> 5;
  const int wm   = (wave & 3) * 32;   // 4 waves along M
  const int wn   = (wave >> 2) * 64;  // 2 waves along N
  const int fr   = lane & 15;         // fragment row/col
  const int kh   = (lane >> 4) * 8;   // half-lane K offset (ISA 7.12.2)

  v8f acc[2][4];
#pragma unroll
  for (int i = 0; i < 2; i++)
#pragma unroll
    for (int j = 0; j < 4; j++)
#pragma unroll
      for (int r = 0; r < 8; r++) acc[i][j][r] = 0.f;

  const int lr = tid >> 1;          // 0..127
  const int lc = (tid & 1) << 4;    // 0 or 16

  for (int k0 = 0; k0 < g.K; k0 += 32) {
    // ---- stage A tile ----
    if constexpr (AHALF) {
      // f16 A, full tiles: pure DMA, 2 x 16B chunks per thread.
      const _Float16* Ah = (const _Float16*)g.A + (long long)z * g.sA;
#pragma unroll
      for (int p = 0; p < 2; p++) {
        const int cid = tid + p * 256;       // 0..511 chunks of 8 halves
        const int row = cid >> 2;
        const int c   = (cid & 3) * 8;
        const _Float16* src = Ah + (long long)(m0 + row) * g.lda + k0 + c;
#if USE_ASYNC_LDS
        async_copy_b128((unsigned)(uintptr_t)&lsA[row * 40 + c], (const void*)src);
#else
        *(v8h*)&lsA[row * 40 + c] = *(const v8h*)src;
#endif
      }
    } else {
      const float* A = (const float*)g.A + (long long)z * g.sA;
      _Float16* dst = &lsA[lr * 40 + lc];
      const int m = m0 + lr;
      if (m < M) {
        const long long row = gat ? (long long)gat[m] : (long long)(rowBase + m);
        const float* srcf = A + row * (long long)g.lda + k0 + lc;
        const float4* s4 = (const float4*)srcf;
        float4 a0 = s4[0], a1 = s4[1], a2 = s4[2], a3 = s4[3];
        if (k0 + 32 < g.K) __builtin_prefetch(srcf + 32, 0, 0);
        v8h h0 = { (_Float16)a0.x, (_Float16)a0.y, (_Float16)a0.z, (_Float16)a0.w,
                   (_Float16)a1.x, (_Float16)a1.y, (_Float16)a1.z, (_Float16)a1.w };
        v8h h1 = { (_Float16)a2.x, (_Float16)a2.y, (_Float16)a2.z, (_Float16)a2.w,
                   (_Float16)a3.x, (_Float16)a3.y, (_Float16)a3.z, (_Float16)a3.w };
        *(v8h*)(dst)     = h0;
        *(v8h*)(dst + 8) = h1;
      } else {
        v8h zz = {};
        *(v8h*)(dst)     = zz;
        *(v8h*)(dst + 8) = zz;
      }
    }
    // ---- stage B tile ----
    if (TRANSB) {
      _Float16* dst = &lsB[lr * 40 + lc];
      const int n = n0 + lr;
      if (n < g.N) {
        const float* srcf = B + (long long)n * g.ldb + k0 + lc;
        const float4* s4 = (const float4*)srcf;
        float4 a0 = s4[0], a1 = s4[1], a2 = s4[2], a3 = s4[3];
        if (k0 + 32 < g.K) __builtin_prefetch(srcf + 32, 0, 0);   // next K chunk
        v8h h0 = { (_Float16)a0.x, (_Float16)a0.y, (_Float16)a0.z, (_Float16)a0.w,
                   (_Float16)a1.x, (_Float16)a1.y, (_Float16)a1.z, (_Float16)a1.w };
        v8h h1 = { (_Float16)a2.x, (_Float16)a2.y, (_Float16)a2.z, (_Float16)a2.w,
                   (_Float16)a3.x, (_Float16)a3.y, (_Float16)a3.z, (_Float16)a3.w };
        *(v8h*)(dst)     = h0;
        *(v8h*)(dst + 8) = h1;
      } else {
        v8h zz = {};
        *(v8h*)(dst)     = zz;
        *(v8h*)(dst + 8) = zz;
      }
    } else {
      // B is [K,N]; transpose-on-load into lsB[n][k]
      const int n  = tid & 127;
      const int ks = (tid >> 7) << 4;
      _Float16* dst = &lsB[n * 40 + ks];
      const int gn = n0 + n;
      if (gn < g.N) {
#pragma unroll
        for (int kk = 0; kk < 16; kk++)
          dst[kk] = (_Float16)B[(long long)(k0 + ks + kk) * g.ldb + gn];
      } else {
#pragma unroll
        for (int kk = 0; kk < 16; kk++) dst[kk] = (_Float16)0.f;
      }
    }
#if USE_ASYNC_LDS
    if constexpr (AHALF) wait_async0();
#endif
    __syncthreads();

    // ---- fragments + WMMA ----
    v16h af[2], bf[4];
#pragma unroll
    for (int i = 0; i < 2; i++) {
      const _Float16* p = &lsA[(wm + i * 16 + fr) * 40 + kh];
      v8h lo = *(const v8h*)p;
      v8h hi = *(const v8h*)(p + 16);
      af[i] = __builtin_shufflevector(lo, hi, 0,1,2,3,4,5,6,7,8,9,10,11,12,13,14,15);
    }
#pragma unroll
    for (int j = 0; j < 4; j++) {
      const _Float16* p = &lsB[(wn + j * 16 + fr) * 40 + kh];
      v8h lo = *(const v8h*)p;
      v8h hi = *(const v8h*)(p + 16);
      bf[j] = __builtin_shufflevector(lo, hi, 0,1,2,3,4,5,6,7,8,9,10,11,12,13,14,15);
    }
#pragma unroll
    for (int i = 0; i < 2; i++)
#pragma unroll
      for (int j = 0; j < 4; j++)
        acc[i][j] = __builtin_amdgcn_wmma_f32_16x16x32_f16(
            false, af[i], false, bf[j], (short)0, acc[i][j], false, false);
    __syncthreads();
  }

  // ---- store C (ISA C layout: VGPR r -> M=r (+8 for lanes 16..31)) ----
#pragma unroll
  for (int i = 0; i < 2; i++)
#pragma unroll
    for (int j = 0; j < 4; j++)
#pragma unroll
      for (int r = 0; r < 8; r++) {
        const int m = m0 + wm + i * 16 + kh + r;
        const int n = n0 + wn + j * 16 + fr;
        if (m < M && n < g.N) {
          float* cp = C + (long long)(rowBase + m) * g.ldc + n;
          const float v = acc[i][j][r];
          *cp = ACC ? (*cp + v) : v;
        }
      }
}

// ---------------- elementwise / helper kernels ----------------

__global__ void k_embed(const int* __restrict__ ids, const float* __restrict__ emb,
                        float* __restrict__ h) {
  const int t = blockIdx.x;
  const long long r = (long long)ids[t] * DMODEL;
  for (int d = threadIdx.x; d < DMODEL; d += 256)
    h[(long long)t * DMODEL + d] = emb[r + d];
}

__global__ void k_rmsnorm(const float* __restrict__ x, const float* __restrict__ w,
                          float* __restrict__ y, int n, int ldx, int ldy) {
  const long long r = blockIdx.x;
  const float* xr = x + r * ldx;
  float* yr = y + r * ldy;
  __shared__ float red[256];
  float s = 0.f;
  for (int i = threadIdx.x; i < n; i += 256) { float v = xr[i]; s += v * v; }
  red[threadIdx.x] = s; __syncthreads();
  for (int k = 128; k > 0; k >>= 1) {
    if ((int)threadIdx.x < k) red[threadIdx.x] += red[threadIdx.x + k];
    __syncthreads();
  }
  const float inv = rsqrtf(red[0] / (float)n + EPS_);
  for (int i = threadIdx.x; i < n; i += 256) yr[i] = w[i] * xr[i] * inv;
}

// rope on x[s, head, off..off+RD): out[i]=x[2i]c - x[2i+1]s ; out[i+32]=x[2i+1]c + x[2i]s
__global__ void k_rope(float* __restrict__ x, int rowStride, int headStride, int off) {
  const int s = blockIdx.x, h = blockIdx.y, i = threadIdx.x;  // 32 lanes, one wave
  float* p = x + (long long)s * rowStride + h * headStride + off;
  const float inv = __powf(10000.f, -2.f * (float)i / (float)RD_);
  const float ang = (float)s * inv;
  const float c = __cosf(ang), sn = __sinf(ang);
  const float x0 = p[2 * i], x1 = p[2 * i + 1];
  p[i]      = x0 * c - x1 * sn;
  p[i + 32] = x1 * c + x0 * sn;
}

__global__ void k_build_kf(const float* __restrict__ kv, const float* __restrict__ ckv,
                           float* __restrict__ kf) {
  const int s = blockIdx.x;
  for (int d = threadIdx.x; d < NHEADS * QHD_; d += 256) {
    const int h = d / QHD_, o = d % QHD_;
    float v;
    if (o < ND_) v = kv[(long long)s * NHEADS * (ND_ + VD_) + h * (ND_ + VD_) + o];
    else         v = ckv[(long long)s * (KVLORA + RD_) + KVLORA + (o - ND_)];
    kf[(long long)s * NHEADS * QHD_ + d] = v;
  }
}

// causal softmax over fp32 scores; emits normalized f16 probs (halves AV traffic).
// 3 passes (max, sum, write) recomputing exp -> zero fp32 writeback.
__global__ void k_softmax(const float* __restrict__ sc, _Float16* __restrict__ ph) {
  const float scale = rsqrtf((float)QHD_);
  const int q = blockIdx.x, h = blockIdx.y;
  const float* row = sc + ((long long)h * S_LEN + q) * S_LEN;
  _Float16* prow   = ph + ((long long)h * S_LEN + q) * S_LEN;
  const int L = q + 1;   // causal
  __shared__ float red[256];
  float mx = -3.4e38f;
  for (int i = threadIdx.x; i < L; i += 256) mx = fmaxf(mx, row[i] * scale);
  red[threadIdx.x] = mx; __syncthreads();
  for (int k = 128; k > 0; k >>= 1) {
    if ((int)threadIdx.x < k) red[threadIdx.x] = fmaxf(red[threadIdx.x], red[threadIdx.x + k]);
    __syncthreads();
  }
  mx = red[0]; __syncthreads();
  float s = 0.f;
  for (int i = threadIdx.x; i < L; i += 256) s += __expf(row[i] * scale - mx);
  red[threadIdx.x] = s; __syncthreads();
  for (int k = 128; k > 0; k >>= 1) {
    if ((int)threadIdx.x < k) red[threadIdx.x] += red[threadIdx.x + k];
    __syncthreads();
  }
  const float inv = 1.f / red[0];
  for (int i = threadIdx.x; i < L; i += 256)
    prow[i] = (_Float16)(__expf(row[i] * scale - mx) * inv);
  for (int i = L + (int)threadIdx.x; i < S_LEN; i += 256) prow[i] = (_Float16)0.f;
}

__global__ void k_silu_mul(float* __restrict__ g, const float* __restrict__ u, long long n) {
  const long long stride = (long long)gridDim.x * blockDim.x;
  for (long long i = blockIdx.x * (long long)blockDim.x + threadIdx.x; i < n; i += stride) {
    const float x = g[i];
    g[i] = x * (1.f / (1.f + __expf(-x))) * u[i];
  }
}

// ---------------- MoE routing ----------------

__global__ void k_router(const float* __restrict__ x, const float* __restrict__ wr,
                         float* __restrict__ rsc) {
  const int t = blockIdx.x;
  const int e = threadIdx.x >> 4;
  const int l = threadIdx.x & 15;
  const float* xt = x + (long long)t * DMODEL;
  const float* we = wr + (long long)e * DMODEL;
  float s = 0.f;
  for (int i = l; i < DMODEL; i += 16) s += xt[i] * we[i];
  __shared__ float red[NEXP][17];
  red[e][l] = s; __syncthreads();
  if (l == 0) {
    float tot = 0.f;
    for (int i = 0; i < 16; i++) tot += red[e][i];
    rsc[t * NEXP + e] = 1.f / (1.f + __expf(-tot));
  }
}

__global__ void k_topk(const float* __restrict__ rsc, const float* __restrict__ bias,
                       int* __restrict__ tidx, float* __restrict__ tw, int* __restrict__ counts) {
  const int t = blockIdx.x * blockDim.x + threadIdx.x;
  if (t >= S_LEN) return;
  float s[NEXP], sb[NEXP];
  for (int e = 0; e < NEXP; e++) { s[e] = rsc[t * NEXP + e]; sb[e] = s[e] + bias[e]; }
  const int GE = NEXP / NGRP;   // 4 experts per group
  float gs[NGRP];
  for (int gI = 0; gI < NGRP; gI++) {
    float m1 = -3.4e38f, m2 = -3.4e38f;
    for (int j = 0; j < GE; j++) {
      const float v = sb[gI * GE + j];
      if (v > m1) { m2 = m1; m1 = v; } else if (v > m2) { m2 = v; }
    }
    gs[gI] = m1 + m2;
  }
  int g1 = 0;
  for (int gI = 1; gI < NGRP; gI++) if (gs[gI] > gs[g1]) g1 = gI;
  int g2 = -1;
  for (int gI = 0; gI < NGRP; gI++) {
    if (gI == g1) continue;
    if (g2 < 0 || gs[gI] > gs[g2]) g2 = gI;
  }
  float mk[NEXP];
  for (int e = 0; e < NEXP; e++) {
    const int gI = e / GE;
    mk[e] = (gI == g1 || gI == g2) ? sb[e] : 0.f;
  }
  bool used[NEXP] = {false};
  int sel[TOPK_]; float wv[TOPK_]; float wsum = 0.f;
  for (int j = 0; j < TOPK_; j++) {
    int best = -1;
    for (int e = 0; e < NEXP; e++) {
      if (used[e]) continue;
      if (best < 0 || mk[e] > mk[best]) best = e;
    }
    used[best] = true; sel[j] = best; wv[j] = s[best]; wsum += wv[j];
  }
  const float invn = RSF_ / (wsum + 1e-20f);
  for (int j = 0; j < TOPK_; j++) {
    tidx[t * TOPK_ + j] = sel[j];
    tw[t * TOPK_ + j]   = wv[j] * invn;
    atomicAdd(&counts[sel[j]], 1);
  }
}

__global__ void k_zero_i32(int* p, int n) {
  const int i = blockIdx.x * blockDim.x + threadIdx.x;
  if (i < n) p[i] = 0;
}

__global__ void k_scan_offsets(const int* __restrict__ counts, int* __restrict__ offs) {
  if (threadIdx.x == 0) {
    int a = 0;
    for (int e = 0; e < NEXP; e++) { offs[e] = a; a += counts[e]; }
    offs[NEXP] = a;   // always S_LEN*TOPK_
  }
}

__global__ void k_assign(const int* __restrict__ tidx, const int* __restrict__ offs,
                         int* __restrict__ cursor, int* __restrict__ tokids,
                         int* __restrict__ apos) {
  const int t = blockIdx.x * blockDim.x + threadIdx.x;
  if (t >= S_LEN) return;
  for (int j = 0; j < TOPK_; j++) {
    const int e = tidx[t * TOPK_ + j];
    const int p = offs[e] + atomicAdd(&cursor[e], 1);
    tokids[p] = t;
    apos[t * TOPK_ + j] = p;
  }
}

__global__ void k_moe_scatter(const float* __restrict__ yg, const float* __restrict__ tw,
                              const int* __restrict__ apos, float* __restrict__ h) {
  const int t = blockIdx.x;
  const int p0 = apos[t * 4 + 0], p1 = apos[t * 4 + 1];
  const int p2 = apos[t * 4 + 2], p3 = apos[t * 4 + 3];
  const float w0 = tw[t * 4 + 0], w1 = tw[t * 4 + 1];
  const float w2 = tw[t * 4 + 2], w3 = tw[t * 4 + 3];
  for (int d = threadIdx.x; d < DMODEL; d += 256) {
    const float v = w0 * yg[(long long)p0 * DMODEL + d] + w1 * yg[(long long)p1 * DMODEL + d]
                  + w2 * yg[(long long)p2 * DMODEL + d] + w3 * yg[(long long)p3 * DMODEL + d];
    h[(long long)t * DMODEL + d] += v;
  }
}

// ---------------- host side ----------------

static void launch_gemm(hipStream_t st,
                        const void* A, int lda, long long sA,
                        const float* B, int ldb, long long sB,
                        float* C, int ldc, long long sC,
                        int M, int N, int K, int batch, bool transB, bool acc,
                        bool aHalf = false,
                        const int* seg = nullptr, const int* gat = nullptr, int maxM = 0) {
  if (!maxM) maxM = M;
  GemmArgs a{A, lda, sA, B, ldb, sB, C, ldc, sC, M, N, K, seg, gat};
  dim3 grid((N + 127) / 128, (maxM + 127) / 128, batch);
  dim3 blk(256);
  if (aHalf) {
    k_gemm<false, false, true><<<grid, blk, 0, st>>>(a);   // AV only
  } else if (transB) {
    if (acc) k_gemm<true, true, false><<<grid, blk, 0, st>>>(a);
    else     k_gemm<true, false, false><<<grid, blk, 0, st>>>(a);
  } else {
    if (acc) k_gemm<false, true, false><<<grid, blk, 0, st>>>(a);
    else     k_gemm<false, false, false><<<grid, blk, 0, st>>>(a);
  }
}

extern "C" void kernel_launch(void* const* d_in, const int* in_sizes, int n_in,
                              void* d_out, int out_size, void* d_ws, size_t ws_size,
                              hipStream_t stream) {
  (void)in_sizes; (void)n_in; (void)out_size; (void)ws_size;

  const int*   ids     = (const int*)  d_in[0];
  const float* embed   = (const float*)d_in[1];
  const float* lm_head = (const float*)d_in[2];
  const float* fnorm   = (const float*)d_in[3];

  struct LayerW {
    const float *ln1, *ln2, *q_a, *q_a_ln, *q_b, *kv_a, *kv_a_ln, *kv_b, *o;
  };
  LayerW lw[2];
  for (int l = 0; l < 2; l++) {
    const int b = 4 + l * 9;
    lw[l].ln1     = (const float*)d_in[b + 0];
    lw[l].ln2     = (const float*)d_in[b + 1];
    lw[l].q_a     = (const float*)d_in[b + 2];
    lw[l].q_a_ln  = (const float*)d_in[b + 3];
    lw[l].q_b     = (const float*)d_in[b + 4];
    lw[l].kv_a    = (const float*)d_in[b + 5];
    lw[l].kv_a_ln = (const float*)d_in[b + 6];
    lw[l].kv_b    = (const float*)d_in[b + 7];
    lw[l].o       = (const float*)d_in[b + 8];
  }
  const float* mlp_gate    = (const float*)d_in[22];
  const float* mlp_up      = (const float*)d_in[23];
  const float* mlp_down    = (const float*)d_in[24];
  const float* router_w    = (const float*)d_in[25];
  const float* router_bias = (const float*)d_in[26];
  const float* exp_gate    = (const float*)d_in[27];
  const float* exp_up      = (const float*)d_in[28];
  const float* exp_down    = (const float*)d_in[29];
  const float* sh_gate     = (const float*)d_in[30];
  const float* sh_up       = (const float*)d_in[31];
  const float* sh_down     = (const float*)d_in[32];

  // -------- workspace carve (bump allocator; unit = fp32 element) --------
  size_t off = 0;
  auto alloc = [&](size_t elems) -> float* {
    float* p = (float*)((char*)d_ws + off);
    off += elems * sizeof(float);
    off = (off + 255) & ~(size_t)255;
    return p;
  };
  float* H   = alloc((size_t)S_LEN * DMODEL);
  float* XN  = alloc((size_t)S_LEN * DMODEL);
  float* QA  = alloc((size_t)S_LEN * QLORA);
  float* Q   = alloc((size_t)S_LEN * NHEADS * QHD_);
  float* CKV = alloc((size_t)S_LEN * (KVLORA + RD_));
  float* KV  = alloc((size_t)S_LEN * NHEADS * (ND_ + VD_));
  float* KF  = alloc((size_t)S_LEN * NHEADS * QHD_);
  float* AO  = alloc((size_t)S_LEN * NHEADS * VD_);
  float* SC  = alloc((size_t)NHEADS * S_LEN * S_LEN);
  _Float16* PH = (_Float16*)alloc((size_t)NHEADS * S_LEN * S_LEN / 2);  // f16 probs
  float* BIG = alloc((size_t)2 * S_LEN * I_FF);          // reused by MLP and MoE
  float* RSC = alloc((size_t)S_LEN * NEXP);
  float* TW  = alloc((size_t)S_LEN * TOPK_);
  int* tidx   = (int*)alloc((size_t)S_LEN * TOPK_);
  int* apos   = (int*)alloc((size_t)S_LEN * TOPK_);
  int* tokids = (int*)alloc((size_t)S_LEN * TOPK_);
  int* counts = (int*)alloc(64);      // [0..15]=counts, [16..31]=cursor, [32..48]=offs
  int* cursor = counts + 16;
  int* offs   = counts + 32;

  // -------- MLA attention block (shared by both layers) --------
  auto attn = [&](const LayerW& w) {
    k_rmsnorm<<<S_LEN, 256, 0, stream>>>(H, w.ln1, XN, DMODEL, DMODEL, DMODEL);
    // q = rms(x @ q_a.T) @ q_b.T
    launch_gemm(stream, XN, DMODEL, 0, w.q_a, DMODEL, 0, QA, QLORA, 0,
                S_LEN, QLORA, DMODEL, 1, true, false);
    k_rmsnorm<<<S_LEN, 256, 0, stream>>>(QA, w.q_a_ln, QA, QLORA, QLORA, QLORA);
    launch_gemm(stream, QA, QLORA, 0, w.q_b, QLORA, 0, Q, NHEADS * QHD_, 0,
                S_LEN, NHEADS * QHD_, QLORA, 1, true, false);
    // ckv = x @ kv_a.T ; rope(k_pe); rms(c); kv = c @ kv_b.T
    launch_gemm(stream, XN, DMODEL, 0, w.kv_a, DMODEL, 0, CKV, KVLORA + RD_, 0,
                S_LEN, KVLORA + RD_, DMODEL, 1, true, false);
    k_rope<<<dim3(S_LEN, 1), 32, 0, stream>>>(CKV, KVLORA + RD_, 0, KVLORA);
    k_rmsnorm<<<S_LEN, 256, 0, stream>>>(CKV, w.kv_a_ln, CKV, KVLORA,
                                         KVLORA + RD_, KVLORA + RD_);
    launch_gemm(stream, CKV, KVLORA + RD_, 0, w.kv_b, KVLORA, 0,
                KV, NHEADS * (ND_ + VD_), 0,
                S_LEN, NHEADS * (ND_ + VD_), KVLORA, 1, true, false);
    // rope(q_pe); assemble kf = [k_nope | bcast(k_pe)]
    k_rope<<<dim3(S_LEN, NHEADS), 32, 0, stream>>>(Q, NHEADS * QHD_, QHD_, ND_);
    k_build_kf<<<S_LEN, 256, 0, stream>>>(KV, CKV, KF);
    // scores[h] = Qf_h @ Kf_h^T  (batched over heads via strides)
    launch_gemm(stream, Q, NHEADS * QHD_, QHD_, KF, NHEADS * QHD_, QHD_,
                SC, S_LEN, (long long)S_LEN * S_LEN,
                S_LEN, S_LEN, QHD_, NHEADS, true, false);
    k_softmax<<<dim3(S_LEN, NHEADS), 256, 0, stream>>>(SC, PH);
    // O[h] = P_h @ V_h ; P is f16, staged via async global->LDS (full tiles)
    launch_gemm(stream, PH, S_LEN, (long long)S_LEN * S_LEN,
                KV + ND_, NHEADS * (ND_ + VD_), ND_ + VD_,
                AO, NHEADS * VD_, VD_,
                S_LEN, VD_, S_LEN, NHEADS, false, false, true);
    // h += O @ o.T
    launch_gemm(stream, AO, NHEADS * VD_, 0, w.o, NHEADS * VD_, 0, H, DMODEL, 0,
                S_LEN, DMODEL, NHEADS * VD_, 1, true, true);
  };

  // ==================== forward ====================
  k_embed<<<S_LEN, 256, 0, stream>>>(ids, embed, H);

  // ---- layer 0: attn + dense MLP ----
  attn(lw[0]);
  k_rmsnorm<<<S_LEN, 256, 0, stream>>>(H, lw[0].ln2, XN, DMODEL, DMODEL, DMODEL);
  {
    float* G = BIG;
    float* U = BIG + (size_t)S_LEN * I_FF;
    launch_gemm(stream, XN, DMODEL, 0, mlp_gate, DMODEL, 0, G, I_FF, 0,
                S_LEN, I_FF, DMODEL, 1, true, false);
    launch_gemm(stream, XN, DMODEL, 0, mlp_up, DMODEL, 0, U, I_FF, 0,
                S_LEN, I_FF, DMODEL, 1, true, false);
    k_silu_mul<<<4096, 256, 0, stream>>>(G, U, (long long)S_LEN * I_FF);
    launch_gemm(stream, G, I_FF, 0, mlp_down, I_FF, 0, H, DMODEL, 0,
                S_LEN, DMODEL, I_FF, 1, true, true);
  }

  // ---- layer 1: attn + MoE ----
  attn(lw[1]);
  k_rmsnorm<<<S_LEN, 256, 0, stream>>>(H, lw[1].ln2, XN, DMODEL, DMODEL, DMODEL);

  // routing
  k_router<<<S_LEN, 256, 0, stream>>>(XN, router_w, RSC);
  k_zero_i32<<<1, 64, 0, stream>>>(counts, 32);
  k_topk<<<(S_LEN + 127) / 128, 128, 0, stream>>>(RSC, router_bias, tidx, TW, counts);
  k_scan_offsets<<<1, 32, 0, stream>>>(counts, offs);
  k_assign<<<(S_LEN + 127) / 128, 128, 0, stream>>>(tidx, offs, cursor, tokids, apos);

  // sparse expert compute on gathered tokens (total assignments = S*TOPK = 4096)
  {
    const size_t NA = (size_t)S_LEN * TOPK_;        // 4096
    float* AG  = BIG;
    float* AU  = BIG + NA * MI_;
    float* YG  = BIG + 2 * NA * MI_;
    float* SHG = YG + NA * DMODEL;
    float* SHU = SHG + (size_t)S_LEN * SI_;

    launch_gemm(stream, XN, DMODEL, 0, exp_gate, DMODEL, (long long)MI_ * DMODEL,
                AG, MI_, 0, S_LEN, MI_, DMODEL, NEXP, true, false, false,
                offs, tokids, S_LEN);
    launch_gemm(stream, XN, DMODEL, 0, exp_up, DMODEL, (long long)MI_ * DMODEL,
                AU, MI_, 0, S_LEN, MI_, DMODEL, NEXP, true, false, false,
                offs, tokids, S_LEN);
    k_silu_mul<<<2048, 256, 0, stream>>>(AG, AU, (long long)NA * MI_);
    launch_gemm(stream, AG, MI_, 0, exp_down, MI_, (long long)DMODEL * MI_,
                YG, DMODEL, 0, S_LEN, DMODEL, MI_, NEXP, true, false, false,
                offs, nullptr, S_LEN);
    k_moe_scatter<<<S_LEN, 256, 0, stream>>>(YG, TW, apos, H);

    // shared expert
    launch_gemm(stream, XN, DMODEL, 0, sh_gate, DMODEL, 0, SHG, SI_, 0,
                S_LEN, SI_, DMODEL, 1, true, false);
    launch_gemm(stream, XN, DMODEL, 0, sh_up, DMODEL, 0, SHU, SI_, 0,
                S_LEN, SI_, DMODEL, 1, true, false);
    k_silu_mul<<<1024, 256, 0, stream>>>(SHG, SHU, (long long)S_LEN * SI_);
    launch_gemm(stream, SHG, SI_, 0, sh_down, SI_, 0, H, DMODEL, 0,
                S_LEN, DMODEL, SI_, 1, true, true);
  }

  // ---- final norm + lm_head ----
  k_rmsnorm<<<S_LEN, 256, 0, stream>>>(H, fnorm, XN, DMODEL, DMODEL, DMODEL);
  launch_gemm(stream, XN, DMODEL, 0, lm_head, DMODEL, 0, (float*)d_out, VOCAB, 0,
              S_LEN, VOCAB, DMODEL, 1, true, false);
}
